// DynamicPatchEmbed_6390911336703
// MI455X (gfx1250) — matrix operand order repaired and checked by reading, not compiled, not run on hardware
//
#include <hip/hip_runtime.h>
#include <hip/hip_bf16.h>
#include <cstdint>

// ---------------------------------------------------------------------------
// DynamicPatchEmbed for MI455X (gfx1250, wave32, WMMA)
// ---------------------------------------------------------------------------

typedef __attribute__((ext_vector_type(16))) __bf16    v16bf;
typedef __attribute__((ext_vector_type(8)))  float     v8f;
typedef __attribute__((ext_vector_type(8)))  unsigned  v8u;

#define IMG   224
#define EMB   768
#define BATCH 16
#define NP    196
#define NSLOT 1364
#define NTOT  261

__device__ __forceinline__ int imin(int a, int b) { return a < b ? a : b; }

// Split 16 f32 into hi/lo bf16 fragments via truncation. hi = trunc16(f);
// the residual f - hi is exactly representable in f32 and lo = trunc16(f-hi)
// captures it, so the pair reproduces f to ~2^-16 relative — no rounding
// arithmetic (bfe/add3 chains) needed in the hot loop.
__device__ __forceinline__ void cvt16_hilo(const float f[16], v16bf& hi, v16bf& lo) {
  v8u uhi, ulo;
#pragma unroll
  for (int i = 0; i < 8; ++i) {
    float f0 = f[2 * i], f1 = f[2 * i + 1];
    unsigned u0 = __float_as_uint(f0);
    unsigned u1 = __float_as_uint(f1);
    unsigned m0 = u0 & 0xFFFF0000u;
    unsigned m1 = u1 & 0xFFFF0000u;
    uhi[i] = (u0 >> 16) | m1;
    float l0 = f0 - __uint_as_float(m0);
    float l1 = f1 - __uint_as_float(m1);
    ulo[i] = (__float_as_uint(l0) >> 16) | (__float_as_uint(l1) & 0xFFFF0000u);
  }
  hi = __builtin_bit_cast(v16bf, uhi);
  lo = __builtin_bit_cast(v16bf, ulo);
}

// ---------------------------------------------------------------------------
// 1) Initialize the split-K (atomic) level regions with the conv bias.
//    Levels 112/56/28 occupy the first 84*768 elements of each batch row.
// ---------------------------------------------------------------------------
__global__ void bias_init_kernel(float* __restrict__ allflat,
                                 const float* __restrict__ b112, const float* __restrict__ b56,
                                 const float* __restrict__ b28) {
  size_t idx = (size_t)blockIdx.x * blockDim.x + threadIdx.x;
  const size_t PER = (size_t)84 * EMB;
  if (idx >= (size_t)BATCH * PER) return;
  size_t b = idx / PER, r = idx % PER;
  const float* bias; int c;
  if      (r < (size_t)4  * EMB) { bias = b112; c = (int)( r                  / 4 ); }
  else if (r < (size_t)20 * EMB) { bias = b56;  c = (int)((r - (size_t)4*EMB) / 16); }
  else                           { bias = b28;  c = (int)((r - (size_t)20*EMB)/ 64); }
  allflat[b * (size_t)NSLOT * EMB + r] = bias[c];
}

// ---------------------------------------------------------------------------
// 2) Score convs (1 out channel) -> sigmoid(+1e-8) coarse grids
// ---------------------------------------------------------------------------
template<int K, int GRID>
__global__ void score_kernel(const float* __restrict__ x, const float* __restrict__ w,
                             const float* __restrict__ bias, float* __restrict__ out) {
  const int Kd = 3 * K * K;
  int g  = blockIdx.x;
  int b  = g / (GRID * GRID);
  int p  = g % (GRID * GRID);
  int ph = p / GRID, pw = p % GRID;
  const float* xb = x + (size_t)b * 3 * IMG * IMG + (size_t)(ph * K) * IMG + (size_t)(pw * K);
  float s = 0.f;
  for (int k = threadIdx.x; k < Kd; k += 256) {
    int ci = k / (K * K); int rem = k - ci * K * K;
    int kh = rem / K;     int kw  = rem - kh * K;
    s += xb[(size_t)ci * IMG * IMG + (size_t)kh * IMG + kw] * w[k];
  }
  __shared__ float red[256];
  red[threadIdx.x] = s; __syncthreads();
  for (int o = 128; o > 0; o >>= 1) {
    if (threadIdx.x < o) red[threadIdx.x] += red[threadIdx.x + o];
    __syncthreads();
  }
  if (threadIdx.x == 0) {
    float v = red[0] + bias[0];
    out[g] = 1.f / (1.f + expf(-v)) + 1e-8f;
  }
}

// ---------------------------------------------------------------------------
// 3) Quadtree selection: one wave32 per batch image (inherently sequential)
// ---------------------------------------------------------------------------
__global__ void select_kernel(const float* __restrict__ sc0, const float* __restrict__ sc1,
                              const float* __restrict__ sc2, const float* __restrict__ sc3,
                              int* __restrict__ fidx_ws, float* __restrict__ fidx_out,
                              float* __restrict__ s_out) {
  int b = blockIdx.x, lane = threadIdx.x;           // 32 threads = 1 wave
  __shared__ float dflat[1280];
  __shared__ float s0[NTOT], s1[NTOT], s2[NTOT], s3[NTOT];
  __shared__ int   msk[NTOT];
  __shared__ int   sel[NP];

  for (int i = lane; i < 1280; i += 32) {           // expand coarse grids to 16x16 x 5
    int l = i >> 8, p = i & 255, y = p >> 4, xq = p & 15;
    float v;
    if      (l == 0) v = sc0[b * 4   + (y >> 3) * 2 + (xq >> 3)];
    else if (l == 1) v = sc1[b * 16  + (y >> 2) * 4 + (xq >> 2)];
    else if (l == 2) v = sc2[b * 64  + (y >> 1) * 8 + (xq >> 1)];
    else if (l == 3) v = sc3[b * 256 +  y * 16      +  xq      ];
    else             v = 0.f;
    dflat[i] = v;
  }
  for (int i = lane; i < NTOT; i += 32) { s0[i]=0.f; s1[i]=0.f; s2[i]=0.f; s3[i]=0.f; msk[i]=0; }
  __syncthreads();
  if (lane == 0) { s0[0] = 1.f; s3[0] = (float)IMG; msk[0] = 1; }
  __syncthreads();

  for (int it = 0; it < 65; ++it) {
    int i = 1 + 4 * it;
    float bv = -1.f; int bi = 0;                    // argmax, lowest index on tie
    for (int j = lane; j < NTOT; j += 32) {
      float v = s0[j] * (float)msk[j];
      if (v > bv) { bv = v; bi = j; }
    }
    for (int o = 16; o > 0; o >>= 1) {
      float ov = __shfl_xor(bv, o, 32);
      int   oi = __shfl_xor(bi, o, 32);
      if (ov > bv || (ov == bv && oi < bi)) { bv = ov; bi = oi; }
    }
    if (lane == 0) {
      int mi = bi; msk[mi] = 0;
      int xp = (int)s1[mi], yp = (int)s2[mi], wp = (int)s3[mi];
      int wc  = wp >> 1;
      int lvl = (wp == 224) ? 4 : (wp == 112) ? 3 : (wp == 56) ? 2 : 1;
      int gap = 1 << (lvl - 1);
      int base = (yp / 14) * 16 + xp / 14 + 256 * (4 - lvl);
      float cs[4] = { dflat[base], dflat[base + gap],
                      dflat[base + gap * 16], dflat[base + gap * 16 + gap] };
      int cx[4] = { xp, xp + wc, xp, xp + wc };
      int cy[4] = { yp, yp, yp + wc, yp + wc };
      int cm = (wc != 7) ? 1 : 0;
      for (int c = 0; c < 4; ++c) {
        s0[i+c] = cs[c]; s1[i+c] = (float)cx[c]; s2[i+c] = (float)cy[c];
        s3[i+c] = (float)wc; msk[i+c] = cm;
      }
    }
    __syncthreads();
  }
  for (int j = lane; j < NTOT; j += 32) if (s3[j] == 7.f) msk[j] = 1;
  __syncthreads();
  if (lane == 0) {                                  // stable extraction of mask==1
    int cnt = 0;
    for (int j = 0; j < NTOT && cnt < NP; ++j) if (msk[j]) sel[cnt++] = j;
  }
  __syncthreads();
  for (int j = lane; j < NP; j += 32) {
    int idx = sel[j];
    int xi = (int)s1[idx], yi = (int)s2[idx], wi = (int)s3[idx];
    int lvl = (wi == 7) ? 0 : (wi == 14) ? 1 : (wi == 28) ? 2 : (wi == 56) ? 3 : 4;
    int l = 4 - lvl;
    int pos = (yi / 14) * 16 + xi / 14;
    float chosen = dflat[l * 256 + pos];
    float parents = 1.f;
    for (int m = 0; m < l; ++m) parents += dflat[m * 256 + pos];
    float S = (1.f - chosen + parents) / (float)(l + 2);
    int wnum = IMG / wi;
    int fi = (wnum * wnum - 4) / 3 + (yi / wi) * wnum + (xi / wi);
    fidx_ws [b * NP + j] = fi;
    fidx_out[b * NP + j] = (float)fi;
    s_out   [b * NP + j] = S;
  }
}

// ---------------------------------------------------------------------------
// 4) Patch-embed conv as WMMA GEMM: out[c,n] = W[768,3K^2] x patches[3K^2, B*P]
//    f32 -> bf16 hi/lo split, 3x v_wmma_f32_16x16x32_bf16 per K-chunk.
//    Unguarded main loop (+ branchless constexpr tail); MT row-subtiles per
//    wave amortize the gathered B fragment; split-K via f32 atomics only for
//    the weight-heavy levels, plain store+bias elsewhere.
// ---------------------------------------------------------------------------
template<int K, int PP, int GRID, int SPLIT, int OFF, int MT, bool ATOMIC>
__global__ __launch_bounds__(32)
void patch_gemm_kernel(const float* __restrict__ x, const float* __restrict__ w,
                       const float* __restrict__ bias, float* __restrict__ allflat) {
  constexpr int Kd = 3 * K * K;
  constexpr int KK = K * K;
  const int lane = threadIdx.x;
  const int hb = lane >> 4, l15 = lane & 15;
  const int m0 = blockIdx.x * (16 * MT);
  const int n0 = blockIdx.y * 16;

  constexpr int nblocks = (Kd + 31) / 32;
  constexpr int cbk = (nblocks + SPLIT - 1) / SPLIT;
  const int kb0 = blockIdx.z * cbk;
  const int kb1 = imin(kb0 + cbk, nblocks);
  const int kbF = imin(kb1, Kd / 32);      // fully-populated 32-k blocks

  // B operand: this lane's column of the patch matrix
  const int col  = n0 + l15;
  const int bimg = col / PP;
  const int p    = col % PP;
  const int ph   = p / GRID, pw = p % GRID;
  const float* xb = x + (size_t)bimg * 3 * IMG * IMG
                      + (size_t)(ph * K) * IMG + (size_t)(pw * K);
  // A operand: this lane's weight rows (one per 16-row subtile)
  const float* wr[MT];
#pragma unroll
  for (int mt = 0; mt < MT; ++mt)
    wr[mt] = w + (size_t)(m0 + mt * 16 + l15) * Kd;

  v8f acc[MT] = {};

  for (int kb = kb0; kb < kbF; ++kb) {
    // ---- B fragment: lanes<16 hold K in {0..7,16..23}, lanes>=16 {8..15,24..31}
    const int kbB = kb * 32 + hb * 8;
    int ci = kbB / KK; int rem = kbB - ci * KK;
    int kh = rem / K;  int kw  = rem - kh * K;
    float bfv[16];
#pragma unroll
    for (int t = 0; t < 16; ++t) {
      if (t == 8) { int k2 = kbB + 16; ci = k2 / KK; rem = k2 - ci * KK; kh = rem / K; kw = rem - kh * K; }
      bfv[t] = xb[(size_t)ci * (IMG * IMG) + (size_t)kh * IMG + kw];
      if (++kw == K) { kw = 0; if (++kh == K) { kh = 0; ++ci; } }
    }
    v16bf bhi, blo;
    cvt16_hilo(bfv, bhi, blo);
    // ---- A fragments: lanes<16 hold K in {0..7,16..23}, lanes>=16 {8..15,24..31}
    const int ka = kb * 32 + hb * 8;
#pragma unroll
    for (int mt = 0; mt < MT; ++mt) {
      __builtin_prefetch((const void*)(wr[mt] + (size_t)(kb + 2) * 32), 0, 1);
      float af[16];
      if constexpr ((Kd & 3) == 0) {                // 16B-aligned vector loads
        const float4* q = (const float4*)(wr[mt] + ka);
        float4 a0 = q[0], a1 = q[1], a2 = q[4], a3 = q[5];
        af[0]=a0.x; af[1]=a0.y; af[2]=a0.z; af[3]=a0.w;
        af[4]=a1.x; af[5]=a1.y; af[6]=a1.z; af[7]=a1.w;
        af[8]=a2.x; af[9]=a2.y; af[10]=a2.z; af[11]=a2.w;
        af[12]=a3.x; af[13]=a3.y; af[14]=a3.z; af[15]=a3.w;
      } else {
#pragma unroll
        for (int t = 0; t < 8; ++t) { af[t] = wr[mt][ka + t]; af[8 + t] = wr[mt][ka + 16 + t]; }
      }
      v16bf ahi, alo;
      cvt16_hilo(af, ahi, alo);
      acc[mt] = __builtin_amdgcn_wmma_f32_16x16x32_bf16(false, ahi, false, bhi, (short)0, acc[mt], false, false);
      acc[mt] = __builtin_amdgcn_wmma_f32_16x16x32_bf16(false, ahi, false, blo, (short)0, acc[mt], false, false);
      acc[mt] = __builtin_amdgcn_wmma_f32_16x16x32_bf16(false, alo, false, bhi, (short)0, acc[mt], false, false);
    }
  }

  // ---- branchless tail block (only instantiated when Kd % 32 != 0)
  if constexpr ((Kd & 31) != 0) {
    if (kb1 > kbF) {
      const int kb = kbF;
      const int kbB = kb * 32 + hb * 8;
      float bfv[16];
#pragma unroll
      for (int t = 0; t < 16; ++t) {
        int k  = kbB + (t < 8 ? t : t + 8);
        int kc = imin(k, Kd - 1);                  // clamp, then select-zero
        int ci = kc / KK; int rem = kc - ci * KK;
        int kh = rem / K; int kw = rem - kh * K;
        float v = xb[(size_t)ci * (IMG * IMG) + (size_t)kh * IMG + kw];
        bfv[t] = (k < Kd) ? v : 0.f;
      }
      v16bf bhi, blo;
      cvt16_hilo(bfv, bhi, blo);
      const int ka = kb * 32 + hb * 8;
#pragma unroll
      for (int mt = 0; mt < MT; ++mt) {
        float af[16];
#pragma unroll
        for (int t = 0; t < 8; ++t) {
          int k1 = ka + t, k2 = ka + 16 + t;
          float v1 = wr[mt][imin(k1, Kd - 1)];
          float v2 = wr[mt][imin(k2, Kd - 1)];
          af[t]     = (k1 < Kd) ? v1 : 0.f;
          af[8 + t] = (k2 < Kd) ? v2 : 0.f;
        }
        v16bf ahi, alo;
        cvt16_hilo(af, ahi, alo);
        acc[mt] = __builtin_amdgcn_wmma_f32_16x16x32_bf16(false, ahi, false, bhi, (short)0, acc[mt], false, false);
        acc[mt] = __builtin_amdgcn_wmma_f32_16x16x32_bf16(false, ahi, false, blo, (short)0, acc[mt], false, false);
        acc[mt] = __builtin_amdgcn_wmma_f32_16x16x32_bf16(false, alo, false, bhi, (short)0, acc[mt], false, false);
      }
    }
  }

  // ---- C/D layout: VGPR r -> M = tile + r + 8*hb, N = n0 + (lane&15)
  float* obase = allflat + (size_t)bimg * NSLOT * EMB + OFF;
#pragma unroll
  for (int mt = 0; mt < MT; ++mt) {
#pragma unroll
    for (int r = 0; r < 8; ++r) {
      int rr = m0 + mt * 16 + hb * 8 + r;
      float* addr = obase + (size_t)rr * PP + p;
      if constexpr (ATOMIC) atomicAdd(addr, acc[mt][r]);
      else                  *addr = acc[mt][r] + bias[rr];
    }
  }
}

// ---------------------------------------------------------------------------
// 5) Gather: emb[b,j,:] = allflat[b, flat_idx[b,j], :]
// ---------------------------------------------------------------------------
__global__ void gather_kernel(const float* __restrict__ allflat, const int* __restrict__ fidx,
                              float* __restrict__ out) {
  int j = blockIdx.x, b = blockIdx.y;
  int slot = fidx[b * NP + j];
  const float4* src = (const float4*)(allflat + ((size_t)b * NSLOT + slot) * EMB);
  float4*       dst = (float4*)(out + ((size_t)b * NP + j) * EMB);
  for (int e = threadIdx.x; e < EMB / 4; e += blockDim.x) dst[e] = src[e];
}

// ---------------------------------------------------------------------------
// Launch
// ---------------------------------------------------------------------------
extern "C" void kernel_launch(void* const* d_in, const int* in_sizes, int n_in,
                              void* d_out, int out_size, void* d_ws, size_t ws_size,
                              hipStream_t stream) {
  const float* x     = (const float*)d_in[0];
  const float* cw112 = (const float*)d_in[1];  const float* cb112 = (const float*)d_in[2];
  const float* cw56  = (const float*)d_in[3];  const float* cb56  = (const float*)d_in[4];
  const float* cw28  = (const float*)d_in[5];  const float* cb28  = (const float*)d_in[6];
  const float* cw14  = (const float*)d_in[7];  const float* cb14  = (const float*)d_in[8];
  const float* pw112 = (const float*)d_in[9];  const float* pb112 = (const float*)d_in[10];
  const float* pw56  = (const float*)d_in[11]; const float* pb56  = (const float*)d_in[12];
  const float* pw28  = (const float*)d_in[13]; const float* pb28  = (const float*)d_in[14];
  const float* pw14  = (const float*)d_in[15]; const float* pb14  = (const float*)d_in[16];
  const float* pw7   = (const float*)d_in[17]; const float* pb7   = (const float*)d_in[18];

  // workspace layout
  float* allflat = (float*)d_ws;                                    // 16*1364*768 f32
  float* sc0 = allflat + (size_t)BATCH * NSLOT * EMB;               // 16*4
  float* sc1 = sc0 + BATCH * 4;                                     // 16*16
  float* sc2 = sc1 + BATCH * 16;                                    // 16*64
  float* sc3 = sc2 + BATCH * 64;                                    // 16*256
  int*   fidx_ws = (int*)(sc3 + BATCH * 256);                       // 16*196

  // output layout: emb | flat_idx | S
  float* emb_out  = (float*)d_out;
  float* fidx_out = emb_out + (size_t)BATCH * NP * EMB;
  float* s_out    = fidx_out + BATCH * NP;

  // 1) bias init of the atomic (split-K) level regions only
  {
    size_t E = (size_t)BATCH * 84 * EMB;
    bias_init_kernel<<<(unsigned)((E + 255) / 256), 256, 0, stream>>>(allflat, pb112, pb56, pb28);
  }
  // 2) score convs
  score_kernel<112, 2> <<<BATCH * 4,   256, 0, stream>>>(x, cw112, cb112, sc0);
  score_kernel<56,  4> <<<BATCH * 16,  256, 0, stream>>>(x, cw56,  cb56,  sc1);
  score_kernel<28,  8> <<<BATCH * 64,  256, 0, stream>>>(x, cw28,  cb28,  sc2);
  score_kernel<14, 16> <<<BATCH * 256, 256, 0, stream>>>(x, cw14,  cb14,  sc3);
  // 3) quadtree selection (also writes flat_idx + S outputs)
  select_kernel<<<BATCH, 32, 0, stream>>>(sc0, sc1, sc2, sc3, fidx_ws, fidx_out, s_out);
  // 4) patch-embed WMMA GEMMs
  patch_gemm_kernel<112, 4,    2,  48, 0,      1, true>  <<<dim3(48, 4,    48), 32, 0, stream>>>(x, pw112, pb112, allflat);
  patch_gemm_kernel<56,  16,   4,  12, 3072,   2, true>  <<<dim3(24, 16,   12), 32, 0, stream>>>(x, pw56,  pb56,  allflat);
  patch_gemm_kernel<28,  64,   8,  3,  15360,  2, true>  <<<dim3(24, 64,   3),  32, 0, stream>>>(x, pw28,  pb28,  allflat);
  patch_gemm_kernel<14,  256,  16, 1,  64512,  2, false> <<<dim3(24, 256,  1),  32, 0, stream>>>(x, pw14,  pb14,  allflat);
  patch_gemm_kernel<7,   1024, 32, 1,  261120, 4, false> <<<dim3(12, 1024, 1),  32, 0, stream>>>(x, pw7,   pb7,   allflat);
  // 5) gather selected embeddings
  gather_kernel<<<dim3(NP, BATCH), 192, 0, stream>>>(allflat, fidx_ws, emb_out);
}
